// MultiheadSelfAttention_43164421325147
// MI455X (gfx1250) — matrix operand order repaired and checked by reading, not compiled, not run on hardware
//
#include <hip/hip_runtime.h>
#include <hip/hip_bf16.h>

typedef _Float16 h16;
typedef __attribute__((ext_vector_type(8)))  _Float16 v8h;
typedef __attribute__((ext_vector_type(16))) _Float16 v16h;
typedef __attribute__((ext_vector_type(8)))  float    v8f;

#define WMMA_F16(a, b, c) \
  __builtin_amdgcn_wmma_f32_16x16x32_f16(false, (a), false, (b), (short)0, (c), false, false)

#define WAIT_ASYNC0() asm volatile("s_wait_asynccnt 0x0" ::: "memory")

// Async DMA: 16 bytes per lane, global -> LDS, tracked by ASYNCcnt.
static __device__ inline void async_ld_b128(void* lds_ptr, const void* gptr) {
  unsigned int l = (unsigned int)(uintptr_t)lds_ptr;  // aperture low 32 bits == LDS byte addr
  asm volatile("global_load_async_to_lds_b128 %0, %1, off"
               :: "v"(l), "v"(gptr) : "memory");
}

static __device__ inline v8f vzero8f() {
  v8f v;
#pragma unroll
  for (int i = 0; i < 8; ++i) v[i] = 0.f;
  return v;
}

// A-fragment (16x32, MxK) per ISA 7.12.2: lane l holds row m=l&15;
// halves j<8 -> K = (l>>4)*8 + j ; j>=8 -> K = 16 + (l>>4)*8 + (j-8).
static __device__ inline v16h load_a_frag(const h16* tile, int ld) {
  const int lane = threadIdx.x & 31;
  const h16* p = tile + (lane & 15) * ld + ((lane >> 4) << 3);
  v8h lo = *reinterpret_cast<const v8h*>(p);
  v8h hi = *reinterpret_cast<const v8h*>(p + 16);
  v16h a;
#pragma unroll
  for (int j = 0; j < 8; ++j) { a[j] = lo[j]; a[j + 8] = hi[j]; }
  return a;
}

// B-fragment (32x16, KxN): lane l holds column n=l&15; halves j<8 -> K = (l>>4)*8 + j ; j>=8 -> K = 16 + (l>>4)*8 + (j-8).
// tile indexed [n][k] row-major with row stride ld.
static __device__ inline v16h load_b_frag(const h16* tile, int ld) {
  const int lane = threadIdx.x & 31;
  const h16* p = tile + (lane & 15) * ld + ((lane >> 4) << 3);
  v8h lo = *reinterpret_cast<const v8h*>(p);
  v8h hi = *reinterpret_cast<const v8h*>(p + 16);
  v16h b;
#pragma unroll
  for (int j = 0; j < 8; ++j) { b[j] = lo[j]; b[j + 8] = hi[j]; }
  return b;
}

__global__ void f32_to_f16_kernel(const float* __restrict__ src, h16* __restrict__ dst, int n) {
  int i = blockIdx.x * 256 + threadIdx.x;
  if (i < n) dst[i] = (h16)src[i];
}

// out[m][n] = sum_k A[m][k] * W[n][k]  (x @ W^T).
// Q,K scattered f16 to [B,H,S,64]; V scattered f16 TRANSPOSED to [B,H,64,S].
__global__ __launch_bounds__(256) void gemm_qkv_kernel(
    const h16* __restrict__ A,
    const h16* __restrict__ Wq, const h16* __restrict__ Wk, const h16* __restrict__ Wv,
    h16* __restrict__ Q, h16* __restrict__ K, h16* __restrict__ V) {
  __shared__ alignas(16) h16 Asub[2][64 * 32];
  __shared__ alignas(16) h16 Bsub[2][128 * 32];
  const int t = threadIdx.x;
  const int m0 = blockIdx.y * 64;
  const int n0 = blockIdx.x * 128;
  const h16* W = (blockIdx.z == 0) ? Wq : ((blockIdx.z == 1) ? Wk : Wv);
  h16* dst = (blockIdx.z == 0) ? Q : ((blockIdx.z == 1) ? K : V);
  const int wave = t >> 5, wm = wave >> 2, wn = wave & 3;

  v8f acc[2][2];
#pragma unroll
  for (int i = 0; i < 2; ++i)
#pragma unroll
    for (int j = 0; j < 2; ++j) acc[i][j] = vzero8f();

  // async tile issue: 1 A-chunk + 2 B-chunks of 16B per thread per tile
  auto issue = [&](int k0, int buf) {
    async_ld_b128(&Asub[buf][(t >> 2) * 32 + ((t & 3) << 3)],
                  A + (m0 + (t >> 2)) * 1024 + k0 + ((t & 3) << 3));
#pragma unroll
    for (int i = 0; i < 2; ++i) {
      int c = t + (i << 8);
      async_ld_b128(&Bsub[buf][(c >> 2) * 32 + ((c & 3) << 3)],
                    W + (n0 + (c >> 2)) * 1024 + k0 + ((c & 3) << 3));
    }
  };

  issue(0, 0);
  for (int it = 0; it < 32; ++it) {
    WAIT_ASYNC0();        // this wave's copies for tile `it` done
    __syncthreads();      // everyone's copies done; everyone done reading buf[(it+1)&1] (iter it-1)
    if (it + 1 < 32) issue((it + 1) * 32, (it + 1) & 1);  // DMA overlaps WMMA below
    const h16* As = Asub[it & 1];
    const h16* Bs = Bsub[it & 1];
    v16h af[2], bf[2];
#pragma unroll
    for (int mi = 0; mi < 2; ++mi) af[mi] = load_a_frag(As + (wm * 32 + mi * 16) * 32, 32);
#pragma unroll
    for (int ni = 0; ni < 2; ++ni) bf[ni] = load_b_frag(Bs + (wn * 32 + ni * 16) * 32, 32);
#pragma unroll
    for (int mi = 0; mi < 2; ++mi)
#pragma unroll
      for (int ni = 0; ni < 2; ++ni) acc[mi][ni] = WMMA_F16(af[mi], bf[ni], acc[mi][ni]);
  }

  const int lane = t & 31, nloc = lane & 15, hg = lane >> 4;
  const bool isV = (blockIdx.z == 2);
#pragma unroll
  for (int mi = 0; mi < 2; ++mi)
#pragma unroll
    for (int ni = 0; ni < 2; ++ni) {
      int col = n0 + wn * 32 + ni * 16 + nloc;
      int h = col >> 6, d = col & 63;
#pragma unroll
      for (int r = 0; r < 8; ++r) {
        int row = m0 + wm * 32 + mi * 16 + r + (hg << 3);
        int b = row >> 11, s = row & 2047;
        if (isV)
          dst[((((b << 4) + h) << 6) + d) * 2048 + s] = (h16)acc[mi][ni][r];
        else
          dst[((((b << 4) + h) << 11) + s) * 64 + d] = (h16)acc[mi][ni][r];
      }
    }
}

__global__ void rope_kernel(h16* __restrict__ Q, h16* __restrict__ K,
                            const int* __restrict__ pos, int npairs) {
  int i = blockIdx.x * 256 + threadIdx.x;
  h16* buf = (i < npairs) ? Q : K;
  if (i >= npairs) i -= npairs;
  int f = i & 31;
  int s = (i >> 5) & 2047;
  int bh = i >> 16;
  float p = (float)pos[s];
  float inv = exp2f(-(float)f * 0.41524101186351843f);  // 10000^(-2f/64)
  float ang = p * inv;
  float cs = cosf(ang), sn = sinf(ang);
  h16* bp = buf + (bh * 2048 + s) * 64 + (f << 1);
  float x1 = (float)bp[0], x2 = (float)bp[1];
  bp[0] = (h16)(x1 * cs - x2 * sn);
  bp[1] = (h16)(x1 * sn + x2 * cs);
}

// Flash attention: grid (32 q-tiles, 32 b*h), 128 threads (4 waves x 16 q rows).
// K in [B,H,S,64], V already transposed in [B,H,64,S] -> both tiles are linear async copies.
__global__ __launch_bounds__(128) void attn_kernel(
    const h16* __restrict__ Q, const h16* __restrict__ K, const h16* __restrict__ Vt_g,
    h16* __restrict__ Out) {
  __shared__ alignas(16) h16 Kt[2][64 * 64];   // [k][d]
  __shared__ alignas(16) h16 Vt[2][64 * 64];   // [d][k]
  __shared__ alignas(16) h16 Pb[4 * 16 * 64];  // per-wave P scratch
  const int t = threadIdx.x;
  const int lane = t & 31, w = t >> 5;
  const int qt = blockIdx.x, bh = blockIdx.y;
  const int b = bh >> 4, h = bh & 15;
  const int q0 = qt * 64 + w * 16;
  const int nloc = lane & 15, hg = lane >> 4;

  const h16* qbase = Q + (bh * 2048 + q0) * 64;
  v16h qf[2];
  qf[0] = load_a_frag(qbase, 64);
  qf[1] = load_a_frag(qbase + 32, 64);

  v8f O[4];
#pragma unroll
  for (int i = 0; i < 4; ++i) O[i] = vzero8f();
  float mrow[8], lrow[8];
#pragma unroll
  for (int r = 0; r < 8; ++r) { mrow[r] = -INFINITY; lrow[r] = 0.f; }

  h16* pw = Pb + w * 16 * 64;
  const h16* kbase  = K + bh * 2048 * 64;   // rows s, 64 halves each (contiguous)
  const h16* vtbase = Vt_g + bh * 64 * 2048; // rows d, stride 2048 halves

  // 4 K-chunks + 4 V-chunks of 16B per thread per tile
  auto issue = [&](int kt, int buf) {
    const int k0 = kt * 64;
#pragma unroll
    for (int i = 0; i < 4; ++i) {
      int c = t + (i << 7);  // 0..511
      async_ld_b128(&Kt[buf][c << 3], kbase + k0 * 64 + (c << 3));
      async_ld_b128(&Vt[buf][c << 3], vtbase + (c >> 3) * 2048 + k0 + ((c & 7) << 3));
    }
  };

  issue(0, 0);
  for (int kt = 0; kt <= qt; ++kt) {
    const int k0 = kt * 64;
    WAIT_ASYNC0();
    __syncthreads();
    if (kt + 1 <= qt) issue(kt + 1, (kt + 1) & 1);  // DMA next tile under compute
    const h16* Kc = Kt[kt & 1];
    const h16* Vc = Vt[kt & 1];

    // S = Q K^T  (4 n-tiles of 16 k-columns)
    v8f S[4];
#pragma unroll
    for (int ni = 0; ni < 4; ++ni) {
      v8f s = vzero8f();
      v16h b0 = load_b_frag(Kc + (ni << 4) * 64, 64);
      v16h b1 = load_b_frag(Kc + (ni << 4) * 64 + 32, 64);
      s = WMMA_F16(qf[0], b0, s);
      s = WMMA_F16(qf[1], b1, s);
      S[ni] = s;
    }

    const bool diag = (kt == qt);
#pragma unroll
    for (int ni = 0; ni < 4; ++ni)
#pragma unroll
      for (int r = 0; r < 8; ++r) {
        float v = S[ni][r] * 0.125f;  // 1/sqrt(64)
        if (diag) {
          int kc = k0 + (ni << 4) + nloc;
          int qr = q0 + r + (hg << 3);
          if (kc > qr) v = -INFINITY;
        }
        S[ni][r] = v;
      }

    // online softmax per row
#pragma unroll
    for (int r = 0; r < 8; ++r) {
      float mx = fmaxf(fmaxf(S[0][r], S[1][r]), fmaxf(S[2][r], S[3][r]));
#pragma unroll
      for (int off = 8; off > 0; off >>= 1) mx = fmaxf(mx, __shfl_xor(mx, off, 16));
      float mn = fmaxf(mrow[r], mx);
      float sc = __expf(mrow[r] - mn);
      float rs = 0.f;
#pragma unroll
      for (int ni = 0; ni < 4; ++ni) {
        float p = __expf(S[ni][r] - mn);
        S[ni][r] = p;
        rs += p;
      }
#pragma unroll
      for (int off = 8; off > 0; off >>= 1) rs += __shfl_xor(rs, off, 16);
      lrow[r] = lrow[r] * sc + rs;
      mrow[r] = mn;
#pragma unroll
      for (int ni = 0; ni < 4; ++ni) O[ni][r] *= sc;
    }

    // C-layout f32 -> A-layout f16 via per-wave LDS (in-order DS within wave)
#pragma unroll
    for (int ni = 0; ni < 4; ++ni)
#pragma unroll
      for (int r = 0; r < 8; ++r)
        pw[(r + (hg << 3)) * 64 + (ni << 4) + nloc] = (h16)S[ni][r];
    asm volatile("s_wait_dscnt 0x0" ::: "memory");

    v16h pa0 = load_a_frag(pw, 64);
    v16h pa1 = load_a_frag(pw + 32, 64);
#pragma unroll
    for (int ni = 0; ni < 4; ++ni) {
      v16h vb0 = load_b_frag(Vc + (ni << 4) * 64, 64);
      v16h vb1 = load_b_frag(Vc + (ni << 4) * 64 + 32, 64);
      O[ni] = WMMA_F16(pa0, vb0, O[ni]);
      O[ni] = WMMA_F16(pa1, vb1, O[ni]);
    }
  }

#pragma unroll
  for (int r = 0; r < 8; ++r) {
    float inv = 1.f / lrow[r];
    int s = q0 + r + (hg << 3);
#pragma unroll
    for (int ni = 0; ni < 4; ++ni) {
      int d = (ni << 4) + nloc;
      Out[(b * 2048 + s) * 1024 + (h << 6) + d] = (h16)(O[ni][r] * inv);
    }
  }
}

// out[m][n] = sum_k A[m][k] * W[n][k], f32 output row-major [4096][1024]
__global__ __launch_bounds__(256) void gemm_out_kernel(
    const h16* __restrict__ A, const h16* __restrict__ W, float* __restrict__ out) {
  __shared__ alignas(16) h16 Asub[2][64 * 32];
  __shared__ alignas(16) h16 Bsub[2][128 * 32];
  const int t = threadIdx.x;
  const int m0 = blockIdx.y * 64;
  const int n0 = blockIdx.x * 128;
  const int wave = t >> 5, wm = wave >> 2, wn = wave & 3;

  v8f acc[2][2];
#pragma unroll
  for (int i = 0; i < 2; ++i)
#pragma unroll
    for (int j = 0; j < 2; ++j) acc[i][j] = vzero8f();

  auto issue = [&](int k0, int buf) {
    async_ld_b128(&Asub[buf][(t >> 2) * 32 + ((t & 3) << 3)],
                  A + (m0 + (t >> 2)) * 1024 + k0 + ((t & 3) << 3));
#pragma unroll
    for (int i = 0; i < 2; ++i) {
      int c = t + (i << 8);
      async_ld_b128(&Bsub[buf][(c >> 2) * 32 + ((c & 3) << 3)],
                    W + (n0 + (c >> 2)) * 1024 + k0 + ((c & 3) << 3));
    }
  };

  issue(0, 0);
  for (int it = 0; it < 32; ++it) {
    WAIT_ASYNC0();
    __syncthreads();
    if (it + 1 < 32) issue((it + 1) * 32, (it + 1) & 1);
    const h16* As = Asub[it & 1];
    const h16* Bs = Bsub[it & 1];
    v16h af[2], bf[2];
#pragma unroll
    for (int mi = 0; mi < 2; ++mi) af[mi] = load_a_frag(As + (wm * 32 + mi * 16) * 32, 32);
#pragma unroll
    for (int ni = 0; ni < 2; ++ni) bf[ni] = load_b_frag(Bs + (wn * 32 + ni * 16) * 32, 32);
#pragma unroll
    for (int mi = 0; mi < 2; ++mi)
#pragma unroll
      for (int ni = 0; ni < 2; ++ni) acc[mi][ni] = WMMA_F16(af[mi], bf[ni], acc[mi][ni]);
  }

  const int lane = t & 31, nloc = lane & 15, hg = lane >> 4;
#pragma unroll
  for (int mi = 0; mi < 2; ++mi)
#pragma unroll
    for (int ni = 0; ni < 2; ++ni) {
      int col = n0 + wn * 32 + ni * 16 + nloc;
#pragma unroll
      for (int r = 0; r < 8; ++r) {
        int row = m0 + wm * 32 + mi * 16 + r + (hg << 3);
        out[row * 1024 + col] = acc[mi][ni][r];
      }
    }
}

extern "C" void kernel_launch(void* const* d_in, const int* in_sizes, int n_in,
                              void* d_out, int out_size, void* d_ws, size_t ws_size,
                              hipStream_t stream) {
  (void)in_sizes; (void)n_in; (void)out_size; (void)ws_size;
  const float* x  = (const float*)d_in[0];
  const int*   tp = (const int*)d_in[1];
  const float* Wq = (const float*)d_in[2];
  const float* Wk = (const float*)d_in[3];
  const float* Wv = (const float*)d_in[4];
  const float* Wo = (const float*)d_in[5];
  float* out = (float*)d_out;

  h16* xh  = (h16*)d_ws;                    // 4096*1024
  h16* wqh = xh  + 4096 * 1024;             // 1024*1024 each
  h16* wkh = wqh + 1024 * 1024;
  h16* wvh = wkh + 1024 * 1024;
  h16* woh = wvh + 1024 * 1024;
  h16* qh  = woh + 1024 * 1024;             // 32*2048*64 each
  h16* kh  = qh  + 32 * 2048 * 64;
  h16* vh  = kh  + 32 * 2048 * 64;          // stored as [B,H,64,S]
  h16* ah  = vh  + 32 * 2048 * 64;          // 4096*1024

  f32_to_f16_kernel<<<16384, 256, 0, stream>>>(x,  xh,  4096 * 1024);
  f32_to_f16_kernel<<<4096,  256, 0, stream>>>(Wq, wqh, 1024 * 1024);
  f32_to_f16_kernel<<<4096,  256, 0, stream>>>(Wk, wkh, 1024 * 1024);
  f32_to_f16_kernel<<<4096,  256, 0, stream>>>(Wv, wvh, 1024 * 1024);
  f32_to_f16_kernel<<<4096,  256, 0, stream>>>(Wo, woh, 1024 * 1024);

  gemm_qkv_kernel<<<dim3(8, 64, 3), 256, 0, stream>>>(xh, wqh, wkh, wvh, qh, kh, vh);
  rope_kernel<<<16384, 256, 0, stream>>>(qh, kh, tp, 2 * 16 * 2048 * 32);
  attn_kernel<<<dim3(32, 32), 128, 0, stream>>>(qh, kh, vh, ah);
  gemm_out_kernel<<<dim3(8, 64), 256, 0, stream>>>(ah, woh, out);
}
